// SimpleSelfAttention_49409303773947
// MI455X (gfx1250) — matrix operand order repaired and checked by reading, not compiled, not run on hardware
//
#include <hip/hip_runtime.h>

// ---------------------------------------------------------------------------
// SimpleSelfAttention for MI455X (gfx1250, wave32, WMMA, TDM).
// B=128, S=512, D=512.  All GEMMs run through v_wmma_f32_16x16x32_bf16 with
// fp32 accumulation; softmax in fp32.  ~172 GFLOP, ~1.3 GB traffic
// (memory roofline ~55us @ 23.3 TB/s; bf16 WMMA compute is comparable).
//
// Pipelined GEMM loop: double-buffered LDS tiles, ONE barrier per K-step.
// Next-tile global loads (VGPR staging) and the next TDM descriptor are
// issued so they overlap the current tile's WMMAs.  TDM tiles use the D#
// pad fields to reproduce the padded LDS row stride (LDK=40) in hardware.
//
// Workspace layout (needs 320 MiB):
//   [0, 64MB)    Qb  bf16  (B,S,D) natural
//   [64,128MB)   Kb  bf16  (B,S,D) natural
//   [128,192MB)  Vb  bf16  (B,S,D) natural
//   [192,320MB)  E   fp32  (B,D,D) energy -> softmaxed in place
// ---------------------------------------------------------------------------

typedef __bf16 bf16;
typedef __attribute__((ext_vector_type(16))) __bf16 v16bf;
typedef __attribute__((ext_vector_type(8)))  __bf16 v8bf;
typedef __attribute__((ext_vector_type(8)))  float  v8f;
typedef __attribute__((ext_vector_type(4)))  float  v4f;
typedef __attribute__((ext_vector_type(4)))  unsigned int u32x4;
typedef __attribute__((ext_vector_type(4)))  int          i32x4;
typedef __attribute__((ext_vector_type(8)))  int          i32x8;

#define LDK 40   // LDS row stride (elements) for 32-wide K tiles, +8 pad

// ---- TDM availability -----------------------------------------------------
#ifndef HAVE_TDM
#if defined(__has_builtin)
#if __has_builtin(__builtin_amdgcn_tensor_load_to_lds) && \
    __has_builtin(__builtin_amdgcn_s_wait_tensorcnt)
#define HAVE_TDM 1
#else
#define HAVE_TDM 0
#endif
#else
#define HAVE_TDM 0
#endif
#endif

#if defined(__HIP_DEVICE_COMPILE__) && !HAVE_TDM
#warning "DEVICE PASS: tensor_load_to_lds builtin NOT available -> manual staging fallback in use"
#endif

__device__ __forceinline__ bf16 f2bf(float f) {
    union { float f; unsigned u; } v; v.f = f;
    unsigned r = v.u + 0x7FFFu + ((v.u >> 16) & 1u);  // round-to-nearest-even
    unsigned short h = (unsigned short)(r >> 16);
    bf16 b; __builtin_memcpy(&b, &h, sizeof(b));
    return b;
}

#if HAVE_TDM
// Issue a TDM load of a 128(rows) x 32(cols) bf16 tile from a row-major
// matrix with row stride 512 elements into LDS with row stride LDK=40
// elements (pad 4 DWORDs after every 16 DWORDs = after each 64B row).
// Per-wave instruction: call from ONE wave only. Tracked by TENSORcnt.
__device__ __forceinline__ void tdm_load_tile128x32(const bf16* gsrc, bf16* ldst) {
    const unsigned lds_off = (unsigned)(unsigned long long)ldst;   // addr[31:0]
    const unsigned long long ga = (unsigned long long)gsrc;

    u32x4 g0;
    g0[0] = 1u;                                   // count=1, user mode, no gather
    g0[1] = lds_off;                              // lds_addr
    g0[2] = (unsigned)(ga & 0xffffffffu);         // global_addr[31:0]
    g0[3] = (unsigned)((ga >> 32) & 0x01ffffffu)  // global_addr[56:32]
          | (2u << 30);                           // type = 2 ("image")

    i32x8 g1;
    g1[0] = (int)((1u << 16)      // data_size = 1 (2 bytes)
                | (1u << 20)      // pad_enable
                | (3u << 22)      // pad_interval: every 16 DWORDs (64B row)
                | (3u << 25));    // pad_amount: 4 DWORDs (16B) -> LDK=40
    g1[1] = (int)(512u << 16);    // tensor_dim0[15:0]=512 in bits[31:16]
    g1[2] = (int)(512u << 16);    // tensor_dim0 hi=0 | tensor_dim1[15:0]=512
    g1[3] = (int)(32u << 16);     // tensor_dim1 hi=0 | tile_dim0=32
    g1[4] = (int)128;             // tile_dim1=128 | tile_dim2=0 (2D tile)
    g1[5] = (int)512;             // tensor_dim0_stride[31:0]=512 elements
    g1[6] = 0;                    // stride0 hi | stride1 lo (unused, 2D)
    g1[7] = 0;

    const i32x4 z4 = {0, 0, 0, 0};
#if defined(__clang_major__) && (__clang_major__ >= 23)
    const i32x8 z8 = {0, 0, 0, 0, 0, 0, 0, 0};
    __builtin_amdgcn_tensor_load_to_lds(g0, g1, z4, z4, z8, 0);
#else
    __builtin_amdgcn_tensor_load_to_lds(g0, g1, z4, z4, 0);
#endif
}
#endif  // HAVE_TDM

// ---- register-staged tile helpers (256 threads per 128x32 tile) -----------
struct RegTileF32 { v4f f0, f1, f2, f3; };
struct RegTileBF  { v8bf h0, h1; };

__device__ __forceinline__ RegTileF32 ldg_f32_tile(const float* __restrict__ src,
                                                   int tid) {
    const int row = tid >> 1;
    const int c   = (tid & 1) * 16;
    const float* s = src + (size_t)row * 512 + c;
    RegTileF32 r;
    r.f0 = *(const v4f*)(s + 0);
    r.f1 = *(const v4f*)(s + 4);
    r.f2 = *(const v4f*)(s + 8);
    r.f3 = *(const v4f*)(s + 12);
    return r;
}

__device__ __forceinline__ void sts_f32_tile(bf16* dst, RegTileF32 r, int tid) {
    const int row = tid >> 1;
    const int c   = (tid & 1) * 16;
    v8f lo = __builtin_shufflevector(r.f0, r.f1, 0, 1, 2, 3, 4, 5, 6, 7);
    v8f hi = __builtin_shufflevector(r.f2, r.f3, 0, 1, 2, 3, 4, 5, 6, 7);
    bf16* d = dst + row * LDK + c;
    *(v8bf*)(d)     = __builtin_convertvector(lo, v8bf);  // v_cvt_pk_bf16_f32
    *(v8bf*)(d + 8) = __builtin_convertvector(hi, v8bf);
}

__device__ __forceinline__ RegTileBF ldg_bf16_tile(const bf16* __restrict__ src,
                                                   int tid) {
    const int row = tid >> 1;
    const int c   = (tid & 1) * 16;
    const bf16* s = src + (size_t)row * 512 + c;
    RegTileBF r;
    r.h0 = *(const v8bf*)(s);
    r.h1 = *(const v8bf*)(s + 8);
    return r;
}

__device__ __forceinline__ void sts_bf16_tile(bf16* dst, RegTileBF r, int tid) {
    const int row = tid >> 1;
    const int c   = (tid & 1) * 16;
    bf16* d = dst + row * LDK + c;
    *(v8bf*)(d)     = r.h0;
    *(v8bf*)(d + 8) = r.h1;
}

// Transposed staging: read a 32x128 bf16 region (row stride 512), emit it as
// a 128x32 LDS tile (forms Q^T tiles from naturally-stored Q).
__device__ __forceinline__ RegTileBF ldg_bf16T_tile(const bf16* __restrict__ src,
                                                    int tid) {
    const int sr = tid >> 3;          // 0..31  source row (contraction index)
    const int c0 = (tid & 7) * 16;    // 0..112 source col (output M index)
    RegTileBF r;
    r.h0 = *(const v8bf*)(src + (size_t)sr * 512 + c0);
    r.h1 = *(const v8bf*)(src + (size_t)sr * 512 + c0 + 8);
    return r;
}

__device__ __forceinline__ void sts_bf16T_tile(bf16* dst, RegTileBF r, int tid) {
    const int sr = tid >> 3;
    const int c0 = (tid & 7) * 16;
#pragma unroll
    for (int i = 0; i < 8; ++i) {
        dst[(c0 + i)     * LDK + sr] = r.h0[i];
        dst[(c0 + 8 + i) * LDK + sr] = r.h1[i];
    }
}

// ---- WMMA fragment loads ---------------------------------------------------
// A-fragment (16x32, MxK). Lane l: row M=(l&15),
// K = {(l>>4)*8 .. +7} U {(l>>4)*8+16 .. +23}.
__device__ __forceinline__ v16bf load_frag_a(const bf16* lds, int row0, int lane) {
    const int r  = row0 + (lane & 15);
    const int k0 = (lane >> 4) * 8;
    const bf16* p = lds + r * LDK + k0;
    v8bf c0 = *(const v8bf*)(p);
    v8bf c1 = *(const v8bf*)(p + 16);
    v16bf a;
#pragma unroll
    for (int i = 0; i < 8; ++i) { a[i] = c0[i]; a[i + 8] = c1[i]; }
    return a;
}

// B-fragment (32x16, KxN) from an NT-staged tile (tile row = N, cols = K).
// Lane l: column N=(l&15), K = {(l>>4)*8 .. +7} U {(l>>4)*8+16 .. +23}.
__device__ __forceinline__ v16bf load_frag_b(const bf16* lds, int col0, int lane) {
    const int r  = col0 + (lane & 15);
    const int k0 = (lane >> 4) * 8;
    const bf16* p = lds + r * LDK + k0;
    v8bf c0 = *(const v8bf*)(p);
    v8bf c1 = *(const v8bf*)(p + 16);
    v16bf b;
#pragma unroll
    for (int i = 0; i < 8; ++i) { b[i] = c0[i]; b[i + 8] = c1[i]; }
    return b;
}

__device__ __forceinline__ v8f wmma_bf16(v16bf a, v16bf b, v8f c) {
    return __builtin_amdgcn_wmma_f32_16x16x32_bf16(
        false, a, false, b, (short)0, c, false, false);
}

#define DECL_ACC(acc)                                     \
    v8f acc[2][4];                                        \
    _Pragma("unroll")                                     \
    for (int mt = 0; mt < 2; ++mt)                        \
        _Pragma("unroll")                                 \
        for (int nt = 0; nt < 4; ++nt)                    \
            _Pragma("unroll")                             \
            for (int v = 0; v < 8; ++v) acc[mt][nt][v] = 0.0f;

#define WMMA_STEP(acc, Acur, Bcur, wm, wn, lane)          \
    {                                                     \
        v16bf a0 = load_frag_a(Acur, wm, lane);           \
        v16bf a1 = load_frag_a(Acur, (wm) + 16, lane);    \
        _Pragma("unroll")                                 \
        for (int nt = 0; nt < 4; ++nt) {                  \
            v16bf bb = load_frag_b(Bcur, (wn) + nt * 16, lane); \
            acc[0][nt] = wmma_bf16(a0, bb, acc[0][nt]);   \
            acc[1][nt] = wmma_bf16(a1, bb, acc[1][nt]);   \
        }                                                 \
    }

// ---------------------------------------------------------------------------
// Kernel 1: Y = x @ W^T + bias -> bf16 natural layout.
// M = B*S = 65536, N = 512, K = 512. Grid (4, 512), 256 threads.
// ---------------------------------------------------------------------------
__global__ __launch_bounds__(256) void proj_kernel(
    const float* __restrict__ x, const float* __restrict__ W,
    const float* __restrict__ bias, bf16* __restrict__ out) {
    __shared__ bf16 As[2][128 * LDK];
    __shared__ bf16 Bs[2][128 * LDK];
    const int tid = threadIdx.x;
    const int m0  = blockIdx.y * 128;
    const int n0  = blockIdx.x * 128;
    const int wave = tid >> 5, lane = tid & 31;
    const int wm = (wave & 3) * 32, wn = (wave >> 2) * 64;
    const float* xb = x + (size_t)m0 * 512;
    const float* Wb = W + (size_t)n0 * 512;

    DECL_ACC(acc)

    RegTileF32 ra = ldg_f32_tile(xb, tid);
    RegTileF32 rb = ldg_f32_tile(Wb, tid);
    int it = 0;
    for (int kt = 0; kt < 512; kt += 32, ++it) {
        sts_f32_tile(&As[it & 1][0], ra, tid);
        sts_f32_tile(&Bs[it & 1][0], rb, tid);
        if (kt + 32 < 512) {                 // next-tile loads fly over WMMAs
            ra = ldg_f32_tile(xb + kt + 32, tid);
            rb = ldg_f32_tile(Wb + kt + 32, tid);
        }
        __syncthreads();                      // single barrier per K-step
        WMMA_STEP(acc, &As[it & 1][0], &Bs[it & 1][0], wm, wn, lane)
    }
#pragma unroll
    for (int mt = 0; mt < 2; ++mt)
#pragma unroll
        for (int nt = 0; nt < 4; ++nt) {
            const int n = n0 + wn + nt * 16 + (lane & 15);
            const float bv = bias[n];
#pragma unroll
            for (int v = 0; v < 8; ++v) {
                const int m = m0 + wm + mt * 16 + v + ((lane >> 4) * 8);
                out[(size_t)m * 512 + n] = f2bf(acc[mt][nt][v] + bv);
            }
        }
}

// ---------------------------------------------------------------------------
// Kernel 2: energy[b,i,j] = sum_s Q[b,s,i] * K[b,j,s] -> fp32.
// A = Q^T (reg-pipelined transpose stage), B = K tile via TDM (double-buffer,
// wait pre-barrier / issue post-barrier so DMA overlaps WMMA). Grid (4,4,128).
// ---------------------------------------------------------------------------
__global__ __launch_bounds__(256) void energy_kernel(
    const bf16* __restrict__ Qb, const bf16* __restrict__ Kb,
    float* __restrict__ E) {
    __shared__ bf16 As[2][128 * LDK];
    __shared__ bf16 Bs[2][128 * LDK];
    const int tid = threadIdx.x;
    const int b   = blockIdx.z;
    const int i0  = blockIdx.y * 128;
    const int j0  = blockIdx.x * 128;
    const int wave = tid >> 5, lane = tid & 31;
    const int wm = (wave & 3) * 32, wn = (wave >> 2) * 64;
    const bf16* Qbase = Qb + (size_t)b * 262144;
    const bf16* Kbase = Kb + (size_t)b * 262144 + (size_t)j0 * 512;

    DECL_ACC(acc)

    RegTileBF ra = ldg_bf16T_tile(Qbase + i0, tid);
#if HAVE_TDM
    if (wave == 0) tdm_load_tile128x32(Kbase, &Bs[0][0]);
#else
    RegTileBF rb = ldg_bf16_tile(Kbase, tid);
#endif
    int it = 0;
    for (int kt = 0; kt < 512; kt += 32, ++it) {
        sts_bf16T_tile(&As[it & 1][0], ra, tid);
#if !HAVE_TDM
        sts_bf16_tile(&Bs[it & 1][0], rb, tid);
#endif
        if (kt + 32 < 512) {
            ra = ldg_bf16T_tile(Qbase + (size_t)(kt + 32) * 512 + i0, tid);
#if !HAVE_TDM
            rb = ldg_bf16_tile(Kbase + kt + 32, tid);
#endif
        }
#if HAVE_TDM
        if (wave == 0) __builtin_amdgcn_s_wait_tensorcnt(0);  // tile `it` landed
#endif
        __syncthreads();                      // single barrier per K-step
#if HAVE_TDM
        if (wave == 0 && kt + 32 < 512)       // old readers of buf done: re-issue
            tdm_load_tile128x32(Kbase + kt + 32, &Bs[(it + 1) & 1][0]);
#endif
        WMMA_STEP(acc, &As[it & 1][0], &Bs[it & 1][0], wm, wn, lane)
    }
    float* Eb = E + (size_t)b * 262144;
#pragma unroll
    for (int mt = 0; mt < 2; ++mt)
#pragma unroll
        for (int nt = 0; nt < 4; ++nt) {
            const int n = j0 + wn + nt * 16 + (lane & 15);
#pragma unroll
            for (int v = 0; v < 8; ++v) {
                const int m = i0 + wm + mt * 16 + v + ((lane >> 4) * 8);
                Eb[(size_t)m * 512 + n] = acc[mt][nt][v];
            }
        }
}

// ---------------------------------------------------------------------------
// Kernel 3: in-place row softmax over (B*512) rows of 512 fp32.
// One wave32 per row, 16 elements per lane.
// ---------------------------------------------------------------------------
__global__ __launch_bounds__(256) void softmax_kernel(float* __restrict__ E) {
    const int row  = blockIdx.x * 8 + (threadIdx.x >> 5);
    const int lane = threadIdx.x & 31;
    float* p = E + (size_t)row * 512;
    float v[16];
    float mx = -3.402823466e38f;
#pragma unroll
    for (int i = 0; i < 16; ++i) {
        v[i] = p[lane + 32 * i];
        mx = fmaxf(mx, v[i]);
    }
#pragma unroll
    for (int off = 16; off >= 1; off >>= 1)
        mx = fmaxf(mx, __shfl_xor(mx, off, 32));
    float sum = 0.0f;
#pragma unroll
    for (int i = 0; i < 16; ++i) {
        v[i] = __expf(v[i] - mx);
        sum += v[i];
    }
#pragma unroll
    for (int off = 16; off >= 1; off >>= 1)
        sum += __shfl_xor(sum, off, 32);
    const float inv = 1.0f / sum;
#pragma unroll
    for (int i = 0; i < 16; ++i)
        p[lane + 32 * i] = v[i] * inv;
}

// ---------------------------------------------------------------------------
// Kernel 4: out[b,s,i] = sum_j attn[b,i,j] * V[b,s,j]   (context^T directly).
// A = V tile via TDM (double-buffered), B = attn rows fp32->bf16 reg-pipelined.
// Output coalesced fp32 (B,S,D). Grid (4, 4, 128).
// ---------------------------------------------------------------------------
__global__ __launch_bounds__(256) void context_kernel(
    const bf16* __restrict__ Vb, const float* __restrict__ E,
    float* __restrict__ out) {
    __shared__ bf16 As[2][128 * LDK];
    __shared__ bf16 Bs[2][128 * LDK];
    const int tid = threadIdx.x;
    const int b   = blockIdx.z;
    const int s0  = blockIdx.y * 128;
    const int i0  = blockIdx.x * 128;
    const int wave = tid >> 5, lane = tid & 31;
    const int wm = (wave & 3) * 32, wn = (wave >> 2) * 64;
    const bf16*  Vbase = Vb + (size_t)b * 262144 + (size_t)s0 * 512;
    const float* Ebase = E  + (size_t)b * 262144 + (size_t)i0 * 512;

    DECL_ACC(acc)

    RegTileF32 rb = ldg_f32_tile(Ebase, tid);
#if HAVE_TDM
    if (wave == 0) tdm_load_tile128x32(Vbase, &As[0][0]);
#else
    RegTileBF ra = ldg_bf16_tile(Vbase, tid);
#endif
    int it = 0;
    for (int kt = 0; kt < 512; kt += 32, ++it) {
        sts_f32_tile(&Bs[it & 1][0], rb, tid);
#if !HAVE_TDM
        sts_bf16_tile(&As[it & 1][0], ra, tid);
#endif
        if (kt + 32 < 512) {
            rb = ldg_f32_tile(Ebase + kt + 32, tid);
#if !HAVE_TDM
            ra = ldg_bf16_tile(Vbase + kt + 32, tid);
#endif
        }
#if HAVE_TDM
        if (wave == 0) __builtin_amdgcn_s_wait_tensorcnt(0);
#endif
        __syncthreads();
#if HAVE_TDM
        if (wave == 0 && kt + 32 < 512)
            tdm_load_tile128x32(Vbase + kt + 32, &As[(it + 1) & 1][0]);
#endif
        WMMA_STEP(acc, &As[it & 1][0], &Bs[it & 1][0], wm, wn, lane)
    }
    float* obase = out + (size_t)b * 262144;
#pragma unroll
    for (int mt = 0; mt < 2; ++mt)
#pragma unroll
        for (int nt = 0; nt < 4; ++nt) {
            const int n = i0 + wn + nt * 16 + (lane & 15);
#pragma unroll
            for (int v = 0; v < 8; ++v) {
                const int m = s0 + wm + mt * 16 + v + ((lane >> 4) * 8);
                obase[(size_t)m * 512 + n] = acc[mt][nt][v];
            }
        }
}

// ---------------------------------------------------------------------------

extern "C" void kernel_launch(void* const* d_in, const int* in_sizes, int n_in,
                              void* d_out, int out_size, void* d_ws, size_t ws_size,
                              hipStream_t stream) {
    const float* x  = (const float*)d_in[0];
    const float* Wq = (const float*)d_in[1];
    const float* bq = (const float*)d_in[2];
    const float* Wk = (const float*)d_in[3];
    const float* bk = (const float*)d_in[4];
    const float* Wv = (const float*)d_in[5];
    const float* bv = (const float*)d_in[6];
    float* out = (float*)d_out;

    char* ws = (char*)d_ws;
    const size_t elems = (size_t)128 * 512 * 512;          // 33,554,432
    bf16*  Qb = (bf16*)(ws);
    bf16*  Kb = (bf16*)(ws + elems * 2);
    bf16*  Vb = (bf16*)(ws + elems * 4);
    float* E  = (float*)(ws + elems * 6);                  // needs 320 MiB total

    dim3 blk(256);
    dim3 gproj(4, 512);          // N-tiles x M-tiles (M = B*S = 65536)
    proj_kernel<<<gproj, blk, 0, stream>>>(x, Wq, bq, Qb);
    proj_kernel<<<gproj, blk, 0, stream>>>(x, Wk, bk, Kb);
    proj_kernel<<<gproj, blk, 0, stream>>>(x, Wv, bv, Vb);

    dim3 ggemm(4, 4, 128);       // j-tiles x i-tiles x batch
    energy_kernel<<<ggemm, blk, 0, stream>>>(Qb, Kb, E);

    softmax_kernel<<<8192, blk, 0, stream>>>(E);           // 65536 rows / 8 per block

    context_kernel<<<ggemm, blk, 0, stream>>>(Vb, E, out);
}
